// DGCNN_11141145166319
// MI455X (gfx1250) — matrix operand order repaired and checked by reading, not compiled, not run on hardware
//
#include <hip/hip_runtime.h>
#include <hip/hip_bf16.h>
#include <math.h>

// ---------------------------------------------------------------------------
// DGCNN forward for MI455X (gfx1250, wave32, WMMA).
// All channel-mixing GEMMs + kNN grams run on v_wmma_f32_16x16x32_f16 with:
//   - padded (mult-of-32) zero-filled operand strides -> unconditional b128 loads
//   - weights pre-packed to B-fragment order in f16   -> 1 contiguous v16h/lane
//   - compile-time 1xNT N-tile register blocking      -> straight-line WMMA runs
//   - software-pipelined A stream (prefetch kc+1 before kc's WMMAs)
// BN+LeakyReLU and the sq_i - 2<xi,xj> + sq_j epilogues are fused in-register.
// ---------------------------------------------------------------------------

typedef __attribute__((ext_vector_type(16))) _Float16 v16h;
typedef __attribute__((ext_vector_type(8)))  float    v8f;
typedef __attribute__((ext_vector_type(4)))  float    v4f;

#define ALPHA 0.2f
#define EPSBN 1e-3f
#define NB    4
#define NPTS  4096
#define KNN   30
#define BNROW (NB * NPTS)       // 16384 point rows
#define BNK   (BNROW * KNN)     // 491520 edge rows

// ---------------------------------------------------------------------------
// Weight packing: f32 [Cin x Cout] -> f16 B-fragment order, zero padded.
// Layout: Wp[((tn*nkc + kc)*32 + lane)*16 + e],  k = kc*32 + (e>=8)*16 + (lane>=16)*8 + (e&7),
//         col = tn*16 + (lane&15).
// ---------------------------------------------------------------------------
__global__ void packW_kernel(const float* __restrict__ W, _Float16* __restrict__ Wp,
                             int Cin, int Cout, int CinP, int CoutP)
{
    int t = blockIdx.x * blockDim.x + threadIdx.x;
    if (t >= CinP * CoutP) return;
    int e    = t & 15;
    int lane = (t >> 4) & 31;
    int rest = t >> 9;
    int nkc  = CinP >> 5;
    int kc = rest % nkc, tn = rest / nkc;
    int k   = kc * 32 + (e >> 3) * 16 + (lane >> 4) * 8 + (e & 7);
    int col = tn * 16 + (lane & 15);
    Wp[t] = (k < Cin && col < Cout) ? (_Float16)W[(size_t)k * Cout + col] : (_Float16)0.f;
}

// ---------------------------------------------------------------------------
// WMMA GEMM: Y[M,ldY] = act(bn(X[M,CinP(padded,zero-tailed)] @ Wp + bias))
// One wave = 16 x (16*NT) strip, NT compile-time. M must be a multiple of 16.
// Pad columns (Cout..CoutP) are stored as 0 so downstream padded reads work.
// flags: bit0 = batchnorm, bit1 = leaky-relu. bias may be nullptr.
// ---------------------------------------------------------------------------
template <int NT>
__global__ __launch_bounds__(128)
void gemm_bn_act_kernel(const float* __restrict__ X, const _Float16* __restrict__ Wp,
                        const float* __restrict__ bias, const float* __restrict__ g,
                        const float* __restrict__ bt, const float* __restrict__ mu,
                        const float* __restrict__ var, float* __restrict__ Y,
                        int M, int CinP, int Cout, int CoutP, int ldY, int flags)
{
    const int wave = threadIdx.x >> 5;
    const int lane = threadIdx.x & 31;
    const int tilesM = M >> 4;
    const int tilesN = CoutP >> 4;
    const int groups = tilesN / NT;
    int wid = blockIdx.x * 4 + wave;
    if (wid >= tilesM * groups) return;            // wave-uniform
    const int tm = wid / groups, gn = wid % groups;
    const int half = lane >> 4, l16 = lane & 15;
    const int nkc  = CinP >> 5;
    const float* xr = X + (size_t)(tm * 16 + l16) * CinP;
    const v16h*  wp = (const v16h*)Wp + (size_t)(gn * NT) * nkc * 32 + lane;

    const v8f zero = {0, 0, 0, 0, 0, 0, 0, 0};
    v8f acc[NT];
#pragma unroll
    for (int nt = 0; nt < NT; ++nt) acc[nt] = zero;

    // software pipeline on the A (activation) stream
    v4f na0 = *(const v4f*)(xr + half * 8);
    v4f na1 = *(const v4f*)(xr + half * 8 + 4);
    v4f na2 = *(const v4f*)(xr + half * 8 + 16);
    v4f na3 = *(const v4f*)(xr + half * 8 + 20);
    for (int kc = 0; kc < nkc; ++kc) {
        v4f a0 = na0, a1 = na1, a2 = na2, a3 = na3;
        if (kc + 1 < nkc) {                         // scalar-uniform branch
            const float* p = xr + (kc + 1) * 32 + half * 8;
            na0 = *(const v4f*)(p);      na1 = *(const v4f*)(p + 4);
            na2 = *(const v4f*)(p + 16); na3 = *(const v4f*)(p + 20);
        }
        v16h bf[NT];
#pragma unroll
        for (int nt = 0; nt < NT; ++nt)             // batched B loads (one clause)
            bf[nt] = wp[(size_t)kc * 32 + (size_t)nt * nkc * 32];
        v16h a;
#pragma unroll
        for (int e = 0; e < 4; ++e) {
            a[e]      = (_Float16)a0[e];
            a[4 + e]  = (_Float16)a1[e];
            a[8 + e]  = (_Float16)a2[e];
            a[12 + e] = (_Float16)a3[e];
        }
#pragma unroll
        for (int nt = 0; nt < NT; ++nt)             // back-to-back WMMAs
            acc[nt] = __builtin_amdgcn_wmma_f32_16x16x32_f16(
                false, a, false, bf[nt], (short)0, acc[nt], false, false);
    }

#pragma unroll
    for (int nt = 0; nt < NT; ++nt) {
        int col = (gn * NT + nt) * 16 + l16;
        if (col >= ldY) continue;                   // store-side only
        bool valid = col < Cout;
        float bb = 0.f, sc = 1.f, sh = 0.f;
        if (valid) {
            bb = bias ? bias[col] : 0.f;
            if (flags & 1) { sc = g[col] * rsqrtf(var[col] + EPSBN); sh = bt[col] - mu[col] * sc; }
        }
#pragma unroll
        for (int v = 0; v < 8; ++v) {
            int r = tm * 16 + v + (half << 3);
            float y = 0.f;
            if (valid) {
                y = acc[nt][v] + bb;
                if (flags & 1) y = y * sc + sh;
                if (flags & 2) y = (y > 0.f) ? y : ALPHA * y;
            }
            Y[(size_t)r * ldY + col] = y;
        }
    }
}

// ---------------------------------------------------------------------------
// kNN: gram via WMMA on padded features, fused distance epilogue, then top-K.
// ---------------------------------------------------------------------------
__global__ void packfeat_kernel(const float* __restrict__ in, float* __restrict__ Xf)
{
    int t = blockIdx.x * blockDim.x + threadIdx.x;
    if (t >= BNROW * 32) return;
    int r = t >> 5, c = t & 31;
    Xf[t] = (c < 6) ? in[(size_t)r * 9 + 3 + c] : 0.f;
}

__global__ void sqnorm_kernel(const float* __restrict__ X, float* __restrict__ sq, int CinP)
{
    int r = blockIdx.x * blockDim.x + threadIdx.x;
    if (r >= BNROW) return;
    const float* xr = X + (size_t)r * CinP;
    float s = 0.f;
    for (int c = 0; c < CinP; ++c) { float v = xr[c]; s += v * v; }
    sq[r] = s;
}

__global__ __launch_bounds__(128)
void dist_wmma_kernel(const float* __restrict__ X, const float* __restrict__ sq,
                      float* __restrict__ Dm, int CinP)
{
    const int wave = threadIdx.x >> 5;
    const int lane = threadIdx.x & 31;
    const int tilesI = NPTS / 16;                  // 256
    const int groupsJ = tilesI / 4;                // 64
    int wid = blockIdx.x * 4 + wave;
    if (wid >= NB * tilesI * groupsJ) return;
    int b  = wid / (tilesI * groupsJ);
    int t  = wid % (tilesI * groupsJ);
    int ti = t / groupsJ, gj = t % groupsJ;
    const int half = lane >> 4, l16 = lane & 15;
    const int nkc = CinP >> 5;
    const float* Xb = X + (size_t)b * NPTS * CinP;
    const float* xr = Xb + (size_t)(ti * 16 + l16) * CinP;

    const v8f zero = {0, 0, 0, 0, 0, 0, 0, 0};
    v8f acc[4];
#pragma unroll
    for (int nt = 0; nt < 4; ++nt) acc[nt] = zero;

    v4f na0 = *(const v4f*)(xr + half * 8);
    v4f na1 = *(const v4f*)(xr + half * 8 + 4);
    v4f na2 = *(const v4f*)(xr + half * 8 + 16);
    v4f na3 = *(const v4f*)(xr + half * 8 + 20);
    for (int kc = 0; kc < nkc; ++kc) {
        v4f a0 = na0, a1 = na1, a2 = na2, a3 = na3;
        if (kc + 1 < nkc) {
            const float* p = xr + (kc + 1) * 32 + half * 8;
            na0 = *(const v4f*)(p);      na1 = *(const v4f*)(p + 4);
            na2 = *(const v4f*)(p + 16); na3 = *(const v4f*)(p + 20);
        }
        // batched B-tile loads: 4 columns strips, two 8-float runs (k = 8h.., 16+8h..) each
        v16h bfr[4];
#pragma unroll
        for (int nt = 0; nt < 4; ++nt) {
            const float* q = Xb + (size_t)((gj * 4 + nt) * 16 + l16) * CinP + kc * 32 + half * 8;
            v4f b0 = *(const v4f*)(q);
            v4f b1 = *(const v4f*)(q + 4);
            v4f b2 = *(const v4f*)(q + 16);
            v4f b3 = *(const v4f*)(q + 20);
#pragma unroll
            for (int e = 0; e < 4; ++e) {
                bfr[nt][e]      = (_Float16)b0[e];
                bfr[nt][4 + e]  = (_Float16)b1[e];
                bfr[nt][8 + e]  = (_Float16)b2[e];
                bfr[nt][12 + e] = (_Float16)b3[e];
            }
        }
        v16h a;
#pragma unroll
        for (int e = 0; e < 4; ++e) {
            a[e]      = (_Float16)a0[e];
            a[4 + e]  = (_Float16)a1[e];
            a[8 + e]  = (_Float16)a2[e];
            a[12 + e] = (_Float16)a3[e];
        }
#pragma unroll
        for (int nt = 0; nt < 4; ++nt)
            acc[nt] = __builtin_amdgcn_wmma_f32_16x16x32_f16(
                false, a, false, bfr[nt], (short)0, acc[nt], false, false);
    }
    const float* sqb = sq + (size_t)b * NPTS;
#pragma unroll
    for (int nt = 0; nt < 4; ++nt) {
        int j = (gj * 4 + nt) * 16 + l16;
        float sqj = sqb[j];
#pragma unroll
        for (int v = 0; v < 8; ++v) {
            int i = ti * 16 + v + (half << 3);
            Dm[((size_t)b * NPTS + i) * NPTS + j] = sqb[i] - 2.f * acc[nt][v] + sqj;
        }
    }
}

// K=30 smallest-distance selection (stable insertion, matches top_k tie order)
__global__ void topk_kernel(const float* __restrict__ Dm, int* __restrict__ idx)
{
    int r = blockIdx.x * blockDim.x + threadIdx.x;
    if (r >= BNROW) return;
    const float* row = Dm + (size_t)r * NPTS;
    float bd[KNN]; int bi[KNN];
    for (int k = 0; k < KNN; ++k) { bd[k] = 3.4e38f; bi[k] = 0; }
    for (int j = 0; j < NPTS; ++j) {
        float d = row[j];
        if (d < bd[KNN - 1]) {
            int p = KNN - 1;
            while (p > 0 && bd[p - 1] > d) { bd[p] = bd[p - 1]; bi[p] = bi[p - 1]; --p; }
            bd[p] = d; bi[p] = j;
        }
    }
    for (int k = 0; k < KNN; ++k) idx[(size_t)r * KNN + k] = bi[k];
}

// edge rows: [central(C) | nb(C) | rel(C) | ||rel||^2 | 0-pad..ldE]
__global__ void edge_kernel(const float* __restrict__ X, const int* __restrict__ idx,
                            float* __restrict__ E, int C, int ld, int ldE)
{
    int t = blockIdx.x * blockDim.x + threadIdx.x;
    if (t >= BNK) return;
    int bn = t / KNN;
    int b  = bn / NPTS;
    int j  = idx[t];
    const float* ctr = X + (size_t)bn * ld;
    const float* nb  = X + ((size_t)b * NPTS + j) * ld;
    float* e = E + (size_t)t * ldE;
    float dist = 0.f;
    for (int c = 0; c < C; ++c) {
        float cc = ctr[c], nn = nb[c], rr = nn - cc;
        e[c] = cc; e[C + c] = nn; e[2 * C + c] = rr;
        dist += rr * rr;
    }
    e[3 * C] = dist;
    for (int c = 3 * C + 1; c < ldE; ++c) e[c] = 0.f;
}

// ---------------------------------------------------------------------------
// AALayer pieces (all 64-stride padded)
// ---------------------------------------------------------------------------
__global__ void lagg_kernel(const float* __restrict__ Xe, const float* __restrict__ aggW,
                            const float* __restrict__ ab, const float* __restrict__ abt,
                            const float* __restrict__ ag, const float* __restrict__ amu,
                            const float* __restrict__ avar, float* __restrict__ lagg)
{
    int t = blockIdx.x * blockDim.x + threadIdx.x;
    if (t >= BNROW * 64) return;
    int bn = t >> 6, of = t & 63;
    if (of >= 60) { lagg[t] = 0.f; return; }
    int j2 = of / 20, o = of % 20;
    float s = 0.f;
    for (int k = 0; k < KNN; ++k) {
        const float* xr = Xe + ((size_t)bn * KNN + k) * 64;
        const float* wr = aggW + ((size_t)(j2 * KNN + k) * 20) * 20 + o;
        for (int i = 0; i < 20; ++i) {
            int m  = j2 * 20 + i;                 // channel-shuffle: xs[m]=x[(m%3)*20+m/3]
            int ch = (m % 3) * 20 + (m / 3);
            s += xr[ch] * wr[(size_t)i * 20];
        }
    }
    s += ab[of];
    float sc = ag[of] * rsqrtf(avar[of] + EPSBN);
    s = (s - amu[of]) * sc + abt[of];
    lagg[t] = (s > 0.f) ? s : ALPHA * s;
}

// softmax over N (axis=1) of S, then l_out = l_agg * att -> agg[:, 60:120]
__global__ __launch_bounds__(256)
void lsoftmax_kernel(const float* __restrict__ S, const float* __restrict__ lagg,
                     float* __restrict__ agg)
{
    __shared__ float red[256];
    int b = blockIdx.x / 60, c = blockIdx.x % 60;
    int tid = threadIdx.x;
    float mx = -3.4e38f;
    for (int n = tid; n < NPTS; n += 256)
        mx = fmaxf(mx, S[((size_t)b * NPTS + n) * 64 + c]);
    red[tid] = mx; __syncthreads();
    for (int s = 128; s > 0; s >>= 1) { if (tid < s) red[tid] = fmaxf(red[tid], red[tid + s]); __syncthreads(); }
    mx = red[0]; __syncthreads();
    float sum = 0.f;
    for (int n = tid; n < NPTS; n += 256)
        sum += __expf(S[((size_t)b * NPTS + n) * 64 + c] - mx);
    red[tid] = sum; __syncthreads();
    for (int s = 128; s > 0; s >>= 1) { if (tid < s) red[tid] += red[tid + s]; __syncthreads(); }
    float inv = 1.f / red[0];
    for (int n = tid; n < NPTS; n += 256) {
        size_t bn = (size_t)b * NPTS + n;
        float att = __expf(S[bn * 64 + c] - mx) * inv;
        agg[bn * 256 + 60 + c] = lagg[bn * 64 + c] * att;
    }
}

// softmax over k of S, f_agg = x*att; sum/max/mean -> agg 0/120/180; pad 240..255
__global__ void fsoftmax_kernel(const float* __restrict__ S, const float* __restrict__ Xe,
                                float* __restrict__ agg)
{
    int t = blockIdx.x * blockDim.x + threadIdx.x;
    if (t >= BNROW * 60) return;
    int bn = t / 60, d = t % 60;
    float mx = -3.4e38f;
    for (int k = 0; k < KNN; ++k)
        mx = fmaxf(mx, S[((size_t)bn * KNN + k) * 64 + d]);
    float sum = 0.f;
    for (int k = 0; k < KNN; ++k)
        sum += __expf(S[((size_t)bn * KNN + k) * 64 + d] - mx);
    float inv = 1.f / sum;
    float sm = 0.f, mxf = -3.4e38f;
    for (int k = 0; k < KNN; ++k) {
        float p  = __expf(S[((size_t)bn * KNN + k) * 64 + d] - mx) * inv;
        float fa = Xe[((size_t)bn * KNN + k) * 64 + d] * p;
        sm += fa; mxf = fmaxf(mxf, fa);
    }
    agg[(size_t)bn * 256 + d]       = sm;
    agg[(size_t)bn * 256 + 120 + d] = mxf;
    agg[(size_t)bn * 256 + 180 + d] = sm / (float)KNN;
    if (d < 16) agg[(size_t)bn * 256 + 240 + d] = 0.f;
}

// ---------------------------------------------------------------------------
// Head helpers
// ---------------------------------------------------------------------------
__global__ void cat3_kernel(const float* __restrict__ n1, const float* __restrict__ n2,
                            const float* __restrict__ n3, float* __restrict__ out)
{
    int t = blockIdx.x * blockDim.x + threadIdx.x;
    if (t >= BNROW * 192) return;
    int bn = t / 192, c = t % 192;
    float v = 0.f;
    if (c < 60)       v = n1[(size_t)bn * 64 + c];
    else if (c < 120) v = n2[(size_t)bn * 64 + c - 60];
    else if (c < 180) v = n3[(size_t)bn * 64 + c - 120];
    out[t] = v;
}

__global__ __launch_bounds__(256)
void gmax_kernel(const float* __restrict__ X7, float* __restrict__ gm)
{
    __shared__ float red[256];
    int b = blockIdx.x / 1024, c = blockIdx.x % 1024;
    int tid = threadIdx.x;
    float mx = -3.4e38f;
    for (int n = tid; n < NPTS; n += 256)
        mx = fmaxf(mx, X7[((size_t)b * NPTS + n) * 1024 + c]);
    red[tid] = mx; __syncthreads();
    for (int s = 128; s > 0; s >>= 1) { if (tid < s) red[tid] = fmaxf(red[tid], red[tid + s]); __syncthreads(); }
    if (tid == 0) gm[(size_t)b * 1024 + c] = red[0];
}

__global__ void cat1204_kernel(const float* __restrict__ gm, const float* __restrict__ n1,
                               const float* __restrict__ n2, const float* __restrict__ n3,
                               float* __restrict__ out)
{
    int t = blockIdx.x * blockDim.x + threadIdx.x;
    if (t >= BNROW * 1216) return;
    int bn = t / 1216, c = t % 1216;
    int b = bn / NPTS;
    float v = 0.f;
    if (c < 1024)      v = gm[(size_t)b * 1024 + c];
    else if (c < 1084) v = n1[(size_t)bn * 64 + (c - 1024)];
    else if (c < 1144) v = n2[(size_t)bn * 64 + (c - 1084)];
    else if (c < 1204) v = n3[(size_t)bn * 64 + (c - 1144)];
    out[t] = v;
}

// ---------------------------------------------------------------------------
// Host side
// ---------------------------------------------------------------------------
struct CBN { const float *W, *b, *bt, *g, *mu, *var; };
static CBN cbn_at(void* const* d_in, int i) {
    CBN c;
    c.W  = (const float*)d_in[i];     c.b  = (const float*)d_in[i + 1];
    c.bt = (const float*)d_in[i + 2]; c.g  = (const float*)d_in[i + 3];
    c.mu = (const float*)d_in[i + 4]; c.var = (const float*)d_in[i + 5];
    return c;
}
struct AAP { const float *aggW, *ab, *abt, *ag, *amu, *avar, *fW, *lW; CBN conv; };
static AAP aa_at(void* const* d_in, int i) {
    AAP a;
    a.aggW = (const float*)d_in[i];     a.ab   = (const float*)d_in[i + 1];
    a.abt  = (const float*)d_in[i + 2]; a.ag   = (const float*)d_in[i + 3];
    a.amu  = (const float*)d_in[i + 4]; a.avar = (const float*)d_in[i + 5];
    a.conv = cbn_at(d_in, i + 6);
    a.fW   = (const float*)d_in[i + 12];
    a.lW   = (const float*)d_in[i + 13];
    return a;
}

extern "C" void kernel_launch(void* const* d_in, const int* in_sizes, int n_in,
                              void* d_out, int out_size, void* d_ws, size_t ws_size,
                              hipStream_t stream)
{
    (void)in_sizes; (void)n_in; (void)out_size; (void)ws_size;
    const float* inputs = (const float*)d_in[0];
    // JAX pytree order (sorted dict keys): aa0,aa1,aa2,conv1,conv2,out1..out5,out7,outL
    AAP aa0 = aa_at(d_in, 1), aa1 = aa_at(d_in, 15), aa2 = aa_at(d_in, 29);
    CBN conv1 = cbn_at(d_in, 43), conv2 = cbn_at(d_in, 49);
    CBN out1 = cbn_at(d_in, 55), out2 = cbn_at(d_in, 61), out3 = cbn_at(d_in, 67);
    CBN out4 = cbn_at(d_in, 73), out5 = cbn_at(d_in, 79), out7 = cbn_at(d_in, 85);
    CBN outL = cbn_at(d_in, 91);

    // ---- workspace carve-up (floats, 256B-aligned slabs) ----
    float* ws = (float*)d_ws;
    size_t o = 0;
    auto alloc = [&](size_t n) { float* p = ws + o; o += (n + 63) & ~(size_t)63; return p; };
    float* Dm     = alloc((size_t)NB * NPTS * NPTS);   // 268 MB distance matrix
    float* sq     = alloc(BNROW);
    int*   idx    = (int*)alloc(BNK);
    float* Xf     = alloc((size_t)BNROW * 32);         // stage-1 padded kNN features
    float* E      = alloc((size_t)BNK * 192);          // edge tensor (padded)
    float* Xa     = alloc((size_t)BNK * 64);
    float* Xb     = alloc((size_t)BNK * 64);
    float* Fs     = alloc((size_t)BNK * 64);           // f-attention scores
    float* laggB  = alloc((size_t)BNROW * 64);
    float* lS     = alloc((size_t)BNROW * 64);         // l-attention scores
    float* aggB   = alloc((size_t)BNROW * 256);
    float* net1   = alloc((size_t)BNROW * 64);
    float* net2   = alloc((size_t)BNROW * 64);
    float* net3   = alloc((size_t)BNROW * 64);
    float* cat180 = alloc((size_t)BNROW * 192);
    float* X7     = alloc((size_t)BNROW * 1024);
    float* gm     = alloc((size_t)NB * 1024);
    float* cat    = alloc((size_t)BNROW * 1216);
    float* c1     = alloc((size_t)BNROW * 512);
    float* c2     = alloc((size_t)BNROW * 256);
    // packed-weight arena (f16)
    _Float16* hp = (_Float16*)(ws + o);
    size_t ho = 0;
    auto halloc = [&](size_t n) { _Float16* p = hp + ho; ho += (n + 127) & ~(size_t)127; return p; };

    auto pack = [&](const float* W, int Cin, int Cout, int CinP, int CoutP) {
        _Float16* Wp = halloc((size_t)CinP * CoutP);
        int tot = CinP * CoutP;
        packW_kernel<<<(tot + 255) / 256, 256, 0, stream>>>(W, Wp, Cin, Cout, CinP, CoutP);
        return (const _Float16*)Wp;
    };
    const _Float16* p_out1  = pack(out1.W, 28, 60, 32, 64);
    const _Float16* p_out2  = pack(out2.W, 60, 60, 64, 64);
    const _Float16* p_out3  = pack(out3.W, 181, 60, 192, 64);
    const _Float16* p_out4  = pack(out4.W, 60, 60, 64, 64);
    const _Float16* p_out5  = pack(out5.W, 181, 60, 192, 64);
    const _Float16* p_aa0c  = pack(aa0.conv.W, 240, 60, 256, 64);
    const _Float16* p_aa1c  = pack(aa1.conv.W, 240, 60, 256, 64);
    const _Float16* p_aa2c  = pack(aa2.conv.W, 240, 60, 256, 64);
    const _Float16* p_aa0l  = pack(aa0.lW, 60, 60, 64, 64);
    const _Float16* p_aa1l  = pack(aa1.lW, 60, 60, 64, 64);
    const _Float16* p_aa2l  = pack(aa2.lW, 60, 60, 64, 64);
    const _Float16* p_aa0f  = pack(aa0.fW, 60, 60, 64, 64);
    const _Float16* p_aa1f  = pack(aa1.fW, 60, 60, 64, 64);
    const _Float16* p_aa2f  = pack(aa2.fW, 60, 60, 64, 64);
    const _Float16* p_out7  = pack(out7.W, 180, 1024, 192, 1024);
    const _Float16* p_conv1 = pack(conv1.W, 1204, 512, 1216, 512);
    const _Float16* p_conv2 = pack(conv2.W, 512, 256, 512, 256);
    const _Float16* p_outL  = pack(outL.W, 256, 12, 256, 16);

    auto gemm = [&](const float* X, const _Float16* Wp, const CBN* p, float* Y,
                    int M, int CinP, int Cout, int CoutP, int ldY, int flags) {
        int tilesM = M >> 4, tilesN = CoutP >> 4;
        const float* bb = p ? p->b : nullptr;
        const float* gg = p ? p->g : nullptr;
        const float* tt = p ? p->bt : nullptr;
        const float* mm = p ? p->mu : nullptr;
        const float* vv = p ? p->var : nullptr;
        if ((tilesN & 3) == 0) {
            int waves = tilesM * (tilesN >> 2);
            gemm_bn_act_kernel<4><<<(waves + 3) / 4, 128, 0, stream>>>(
                X, Wp, bb, gg, tt, mm, vv, Y, M, CinP, Cout, CoutP, ldY, flags);
        } else {
            int waves = tilesM * tilesN;
            gemm_bn_act_kernel<1><<<(waves + 3) / 4, 128, 0, stream>>>(
                X, Wp, bb, gg, tt, mm, vv, Y, M, CinP, Cout, CoutP, ldY, flags);
        }
    };
    auto knn = [&](const float* Xp, int CinP) {
        sqnorm_kernel<<<(BNROW + 255) / 256, 256, 0, stream>>>(Xp, sq, CinP);
        int waves = NB * 256 * 64;
        dist_wmma_kernel<<<(waves + 3) / 4, 128, 0, stream>>>(Xp, sq, Dm, CinP);
        topk_kernel<<<(BNROW + 127) / 128, 128, 0, stream>>>(Dm, idx);
    };
    auto aal = [&](const float* x /*BNK x 64*/, const AAP& p,
                   const _Float16* pc, const _Float16* pl, const _Float16* pf, float* net) {
        lagg_kernel<<<(BNROW * 64 + 255) / 256, 256, 0, stream>>>(
            x, p.aggW, p.ab, p.abt, p.ag, p.amu, p.avar, laggB);
        gemm(laggB, pl, nullptr, lS, BNROW, 64, 60, 64, 64, 0);
        lsoftmax_kernel<<<NB * 60, 256, 0, stream>>>(lS, laggB, aggB);
        gemm(x, pf, nullptr, Fs, BNK, 64, 60, 64, 64, 0);
        fsoftmax_kernel<<<(BNROW * 60 + 255) / 256, 256, 0, stream>>>(Fs, x, aggB);
        gemm(aggB, pc, &p.conv, net, BNROW, 256, 60, 64, 64, 3);
    };

    // ---- Stage 1 ----
    packfeat_kernel<<<(BNROW * 32 + 255) / 256, 256, 0, stream>>>(inputs, Xf);
    knn(Xf, 32);
    edge_kernel<<<(BNK + 255) / 256, 256, 0, stream>>>(inputs, idx, E, 9, 9, 32);
    gemm(E,  p_out1, &out1, Xa, BNK, 32, 60, 64, 64, 3);
    gemm(Xa, p_out2, &out2, Xb, BNK, 64, 60, 64, 64, 3);
    aal(Xb, aa0, p_aa0c, p_aa0l, p_aa0f, net1);

    // ---- Stage 2 ----
    knn(net1, 64);
    edge_kernel<<<(BNK + 255) / 256, 256, 0, stream>>>(net1, idx, E, 60, 64, 192);
    gemm(E,  p_out3, &out3, Xa, BNK, 192, 60, 64, 64, 3);
    gemm(Xa, p_out4, &out4, Xb, BNK, 64, 60, 64, 64, 3);
    aal(Xb, aa1, p_aa1c, p_aa1l, p_aa1f, net2);

    // ---- Stage 3 ----
    knn(net2, 64);
    edge_kernel<<<(BNK + 255) / 256, 256, 0, stream>>>(net2, idx, E, 60, 64, 192);
    gemm(E, p_out5, &out5, Xa, BNK, 192, 60, 64, 64, 3);
    aal(Xa, aa2, p_aa2c, p_aa2l, p_aa2f, net3);

    // ---- Head ----
    cat3_kernel<<<(BNROW * 192 + 255) / 256, 256, 0, stream>>>(net1, net2, net3, cat180);
    gemm(cat180, p_out7, &out7, X7, BNROW, 192, 1024, 1024, 1024, 3);
    gmax_kernel<<<NB * 1024, 256, 0, stream>>>(X7, gm);
    cat1204_kernel<<<(BNROW * 1216 + 255) / 256, 256, 0, stream>>>(gm, net1, net2, net3, cat);
    gemm(cat, p_conv1, &conv1, c1, BNROW, 1216, 512, 512, 512, 3);
    gemm(c1, p_conv2, &conv2, c2, BNROW, 512, 256, 256, 256, 3);
    gemm(c2, p_outL, &outL, (float*)d_out, BNROW, 256, 12, 16, 12, 0);  // no BN/act
}
